// CrossAttention_37460704756386
// MI455X (gfx1250) — matrix operand order repaired and checked by reading, not compiled, not run on hardware
//
#include <hip/hip_runtime.h>
#include <hip/hip_bf16.h>

typedef __attribute__((ext_vector_type(16))) __bf16 v16bf;
typedef __attribute__((ext_vector_type(8)))  float  v8f;
typedef unsigned int   u32;
typedef unsigned short u16;

union V16 { v16bf v; u32 w[8]; };
union V8  { v8f   v; float f[8]; };

__device__ __forceinline__ u16 f2bf(float f) {
  u32 x = __float_as_uint(f);
  return (u16)((x + 0x7FFFu + ((x >> 16) & 1u)) >> 16);   // RNE
}
__device__ __forceinline__ v8f zero8() { v8f z = {}; return z; }

__device__ __forceinline__ v8f wmma_bf16(v16bf a, v16bf b, v8f c) {
  // (neg_a, A, neg_b, B, c_mod, C, reuse_a, reuse_b)
  return __builtin_amdgcn_wmma_f32_16x16x32_bf16(false, a, false, b, (short)0, c,
                                                 false, false);
}

enum { MODE_QPROJ = 0, MODE_KFOLD = 1, MODE_VFOLD = 2, MODE_OPROJ = 3 };

// C[64x64 tile] = A[M,K] * B[K,N] + bias.  128 threads = 4 waves,
// each wave computes a 16x64 strip as 4x v_wmma_f32_16x16x32_bf16 per K-step.
template <int MODE>
__global__ __launch_bounds__(128) void gemm_wmma(
    const void* __restrict__ Aptr, const float* __restrict__ Bw,
    const float* __restrict__ bias, void* __restrict__ outp,
    int M, int Ktot, int Ntot, int lda, int ldb, int ldc)
{
  __shared__ u16 As[64 * 32];   // [m][k] bf16
  __shared__ u16 Bs[64 * 32];   // [n][k] bf16 (transposed for pair reads)

  const int tid  = threadIdx.x;
  const int wv   = tid >> 5, lane = tid & 31, lmod = lane & 15;
  const int m0   = blockIdx.x * 64, n0 = blockIdx.y * 64;

  v8f acc[4];
  #pragma unroll
  for (int i = 0; i < 4; ++i) acc[i] = zero8();

  for (int k0 = 0; k0 < Ktot; k0 += 32) {
    // ---- stage A (64x32) and B (32x64, transposed) into LDS as bf16 ----
    #pragma unroll
    for (int it = 0; it < 16; ++it) {
      int idx = it * 128 + tid;             // 0..2047
      int ar = idx >> 5, ac = idx & 31;     // A row/col
      u16 av;
      if (MODE == MODE_OPROJ)
        av = ((const u16*)Aptr)[(size_t)(m0 + ar) * lda + (k0 + ac)];
      else
        av = f2bf(((const float*)Aptr)[(size_t)(m0 + ar) * lda + (k0 + ac)]);
      As[ar * 32 + ac] = av;

      int br = idx >> 6, bc = idx & 63;     // B k-row / n-col
      float bvv;
      if (MODE == MODE_KFOLD || MODE == MODE_VFOLD) {
        bvv = 0.f;                          // fold group axis into the weight
        #pragma unroll
        for (int g = 0; g < 4; ++g)
          bvv += Bw[(size_t)(k0 + br) * ldb + g * 64 + (n0 + bc)];
      } else {
        bvv = Bw[(size_t)(k0 + br) * ldb + (n0 + bc)];
      }
      Bs[bc * 32 + br] = f2bf(bvv);
    }
    __syncthreads();

    // ---- A fragment: lanes 0-15 rows M, K 0-7/16-23; lanes 16-31 K 8-15/24-31
    V16 afr;
    const int arow  = wv * 16 + lmod;
    const int koffA = (lane < 16) ? 0 : 8;
    #pragma unroll
    for (int j = 0; j < 8; ++j) {
      int K0 = 2 * j + ((j < 4) ? 0 : 8) + koffA;
      afr.w[j] = *(const u32*)&As[arow * 32 + K0];
    }
    const int koffB = (lane < 16) ? 0 : 8;
    #pragma unroll
    for (int nt = 0; nt < 4; ++nt) {
      V16 bfr;
      int bn = nt * 16 + lmod;              // B lane = column N, elems = K
      #pragma unroll
      for (int j = 0; j < 8; ++j)
        bfr.w[j] = *(const u32*)&Bs[bn * 32 + koffB + 2 * j + ((j < 4) ? 0 : 8)];
      acc[nt] = wmma_bf16(afr.v, bfr.v, acc[nt]);
    }
    __syncthreads();
  }

  // ---- epilogue: bias + store (C layout: VGPR r -> row r + (lane<16?0:8)) ----
  #pragma unroll
  for (int nt = 0; nt < 4; ++nt) {
    int col = n0 + nt * 16 + lmod;
    float bv;
    if (MODE == MODE_KFOLD || MODE == MODE_VFOLD) {
      bv = 0.f;
      #pragma unroll
      for (int g = 0; g < 4; ++g) bv += bias[g * 64 + col];
    } else {
      bv = bias[col];
    }
    V8 a8; a8.v = acc[nt];
    #pragma unroll
    for (int r = 0; r < 8; ++r) {
      int row = m0 + wv * 16 + r + ((lane < 16) ? 0 : 8);
      float val = a8.f[r] + bv;
      if (MODE == MODE_OPROJ)
        ((float*)outp)[(size_t)row * ldc + col] = val;
      else if (MODE == MODE_VFOLD)   // store transposed: [b][d][V]
        ((u16*)outp)[(size_t)((row >> 12) * 64 + col) * 4096 + (row & 4095)] = f2bf(val);
      else
        ((u16*)outp)[(size_t)row * ldc + col] = f2bf(val);
    }
  }
}

// Flash attention: grid (T/64, H, B), 4 waves, each wave = 16 query rows.
// K/V operands are contiguous 32B v16bf global loads (they live in L2: 1MB each).
__global__ __launch_bounds__(128) void attn_wmma(
    const u16* __restrict__ qb, const u16* __restrict__ keff,
    const u16* __restrict__ vefft, u16* __restrict__ outO)
{
  constexpr int T = 4096, V = 4096;
  __shared__ u16 Ps[4][16 * 64];            // per-wave P transpose staging

  const int tid  = threadIdx.x;
  const int wv   = tid >> 5, lane = tid & 31, lmod = lane & 15;
  const int half8 = (lane < 16) ? 0 : 8;
  const int koffA = (lane < 16) ? 0 : 16;
  const int koffB = (lane < 16) ? 0 : 16;
  const int b = blockIdx.z, h = blockIdx.y;
  const int t0 = blockIdx.x * 64 + wv * 16;

  // Q fragments (16x64 bf16 -> two K=32 fragments), loaded once
  V16 qf[2];
  {
    const u16* qbase = qb + ((size_t)(b * T + t0 + lmod)) * 1024 + h * 64;
    #pragma unroll
    for (int f = 0; f < 2; ++f) {
      #pragma unroll
      for (int j = 0; j < 8; ++j) {
        int K0 = f * 32 + 2 * j + koffA;
        qf[f].w[j] = *(const u32*)&qbase[K0];
      }
    }
  }

  v8f oacc[4];
  #pragma unroll
  for (int i = 0; i < 4; ++i) oacc[i] = zero8();
  float mrow[8], lrow[8];
  #pragma unroll
  for (int r = 0; r < 8; ++r) { mrow[r] = -1e30f; lrow[r] = 0.f; }

  for (int v0 = 0; v0 < V; v0 += 64) {
    // ---- S = Q * Keff^T / 8 for 4 column tiles ----
    float sbuf[4][8], rmax[8];
    #pragma unroll
    for (int r = 0; r < 8; ++r) rmax[r] = -1e30f;
    #pragma unroll
    for (int nt = 0; nt < 4; ++nt) {
      const u16* kb = keff + ((size_t)(b * V + v0 + nt * 16 + lmod)) * 64 + koffB;
      v16bf kf0 = *(const v16bf*)(kb);
      v16bf kf1 = *(const v16bf*)(kb + 32);
      v8f s = zero8();
      s = wmma_bf16(qf[0].v, kf0, s);
      s = wmma_bf16(qf[1].v, kf1, s);
      s = s * 0.125f;
      V8 s8; s8.v = s;
      #pragma unroll
      for (int r = 0; r < 8; ++r) {
        sbuf[nt][r] = s8.f[r];
        rmax[r] = fmaxf(rmax[r], s8.f[r]);
      }
    }
    // row max across the 16 lanes sharing each row (halves stay disjoint)
    #pragma unroll
    for (int r = 0; r < 8; ++r)
      #pragma unroll
      for (int mk = 1; mk < 16; mk <<= 1)
        rmax[r] = fmaxf(rmax[r], __shfl_xor(rmax[r], mk, 32));

    float scal[8], rs[8];
    #pragma unroll
    for (int r = 0; r < 8; ++r) {
      float mn = fmaxf(mrow[r], rmax[r]);
      scal[r]  = __expf(mrow[r] - mn);
      mrow[r]  = mn;
      rs[r]    = 0.f;
    }
    // P = exp(S - m), write bf16 to per-wave LDS tile (C layout -> [row][col])
    #pragma unroll
    for (int nt = 0; nt < 4; ++nt)
      #pragma unroll
      for (int r = 0; r < 8; ++r) {
        float p = __expf(sbuf[nt][r] - mrow[r]);
        rs[r] += p;
        Ps[wv][(r + half8) * 64 + nt * 16 + lmod] = f2bf(p);
      }
    #pragma unroll
    for (int r = 0; r < 8; ++r) {
      #pragma unroll
      for (int mk = 1; mk < 16; mk <<= 1)
        rs[r] += __shfl_xor(rs[r], mk, 32);
      lrow[r] = lrow[r] * scal[r] + rs[r];
    }
    // rescale running O
    #pragma unroll
    for (int nt = 0; nt < 4; ++nt) {
      V8 t8; t8.v = oacc[nt];
      #pragma unroll
      for (int r = 0; r < 8; ++r) t8.f[r] *= scal[r];
      oacc[nt] = t8.v;
    }
    // P fragments back from LDS in A layout (DS in-order per wave: no barrier)
    V16 pf[2];
    #pragma unroll
    for (int f = 0; f < 2; ++f)
      #pragma unroll
      for (int j = 0; j < 8; ++j) {
        int K0 = f * 32 + 2 * j + koffA;
        pf[f].w[j] = *(const u32*)&Ps[wv][lmod * 64 + K0];
      }
    // O += P * Veff  (VeffT[b][d][V] -> contiguous K loads per lane)
    #pragma unroll
    for (int nt = 0; nt < 4; ++nt) {
      const u16* vb = vefft + ((size_t)(b * 64 + nt * 16 + lmod)) * V + v0 + koffB;
      v16bf vf0 = *(const v16bf*)(vb);
      v16bf vf1 = *(const v16bf*)(vb + 32);
      oacc[nt] = wmma_bf16(pf[0].v, vf0, oacc[nt]);
      oacc[nt] = wmma_bf16(pf[1].v, vf1, oacc[nt]);
    }
  }

  // ---- O /= l, store bf16 [B,T,H*dh] for the output projection ----
  #pragma unroll
  for (int nt = 0; nt < 4; ++nt) {
    V8 t8; t8.v = oacc[nt];
    #pragma unroll
    for (int r = 0; r < 8; ++r) {
      float o = t8.f[r] / lrow[r];
      int t = t0 + r + half8;
      outO[((size_t)(b * T + t)) * 1024 + h * 64 + nt * 16 + lmod] = f2bf(o);
    }
  }
}

extern "C" void kernel_launch(void* const* d_in, const int* in_sizes, int n_in,
                              void* d_out, int out_size, void* d_ws, size_t ws_size,
                              hipStream_t stream) {
  const float* x   = (const float*)d_in[0];
  const float* adj = (const float*)d_in[1];
  const float* Wq  = (const float*)d_in[2];
  const float* bq  = (const float*)d_in[3];
  const float* Wk  = (const float*)d_in[4];
  const float* bk  = (const float*)d_in[5];
  const float* Wv  = (const float*)d_in[6];
  const float* bv  = (const float*)d_in[7];
  const float* Wo  = (const float*)d_in[8];
  const float* bo  = (const float*)d_in[9];

  // workspace: q_bf16 [8192,1024] | Keff [8192,64] | VeffT [2,64,4096] | O_bf16 [8192,1024]
  u16* qbuf  = (u16*)d_ws;
  u16* keff  = qbuf  + (size_t)2 * 4096 * 1024;
  u16* vefft = keff  + (size_t)2 * 4096 * 64;
  u16* obuf  = vefft + (size_t)2 * 64 * 4096;

  const int M = 2 * 4096;
  dim3 blk(128);

  gemm_wmma<MODE_QPROJ><<<dim3(M / 64, 1024 / 64), blk, 0, stream>>>(
      x, Wq, bq, qbuf, M, 1024, 1024, 1024, 1024, 1024);
  gemm_wmma<MODE_KFOLD><<<dim3(M / 64, 1), blk, 0, stream>>>(
      adj, Wk, bk, keff, M, 1024, 64, 1024, 256, 64);
  gemm_wmma<MODE_VFOLD><<<dim3(M / 64, 1), blk, 0, stream>>>(
      adj, Wv, bv, vefft, M, 1024, 64, 1024, 256, 64);
  attn_wmma<<<dim3(4096 / 64, 16, 2), blk, 0, stream>>>(qbuf, keff, vefft, obuf);
  gemm_wmma<MODE_OPROJ><<<dim3(M / 64, 1024 / 64), blk, 0, stream>>>(
      obuf, Wo, bo, (float*)d_out, M, 1024, 1024, 1024, 1024, 1024);
}
